// MAE_51488067944741
// MI455X (gfx1250) — matrix operand order repaired and checked by reading, not compiled, not run on hardware
//
#include <hip/hip_runtime.h>
#include <hip/hip_bf16.h>
#include <math.h>

// ---------------------------------------------------------------------------
// MAE forward on MI455X (gfx1250, wave32, WMMA).
// GEMMs: f32 -> f16 fragments staged in LDS in WMMA lane order ->
//        v_wmma_f32_16x16x32_f16 (f32 accumulate), ds_load_b128 fragment reads.
// All GEMM shapes here have M % 64 == 0, N % 64 == 0, K % 32 == 0 (no guards).
// Reductions (BN/LN/softmax/loss): f32, wave32 shfl trees, deterministic.
// ---------------------------------------------------------------------------

typedef __attribute__((ext_vector_type(16))) _Float16 v16h;
typedef __attribute__((ext_vector_type(8)))  _Float16 v8h;
typedef __attribute__((ext_vector_type(8)))  float    v8f;

#define NB   128          // batch
#define NP   512          // patches per sample
#define NCIN 256
#define NENH 256
#define NEND 128
#define NV   128          // visible per sample
#define NDD  256          // decoder dim
#define NHEADS 4
#define NDH  64
#define NTOK (NB*NP)      // 65536
#define NVT  (NB*NV)      // 16384
#define EPSV 1e-5f

__device__ inline v8f v8f_zero() {
  v8f z;
#pragma unroll
  for (int i = 0; i < 8; ++i) z[i] = 0.f;
  return z;
}

__device__ inline v8f wmma16x16x32(v16h a, v16h b, v8f c) {
  // neg_a, A, neg_b, B, c_mod, C, reuse_a, reuse_b
  return __builtin_amdgcn_wmma_f32_16x16x32_f16(false, a, false, b, (short)0, c,
                                                false, false);
}

__device__ inline v16h cat8(v8h lo, v8h hi) {
  return __builtin_shufflevector(lo, hi, 0, 1, 2, 3, 4, 5, 6, 7, 8, 9, 10, 11,
                                 12, 13, 14, 15);
}

// fragment = 16 halves stored contiguously per lane (16B-aligned) -> 2x ds_load_b128
__device__ inline v16h load_frag(const _Float16* p) {
  v8h lo = *(const v8h*)p;
  v8h hi = *(const v8h*)(p + 8);
  return cat8(lo, hi);
}

__device__ inline float wave_sum(float v) {
#pragma unroll
  for (int off = 16; off; off >>= 1) v += __shfl_xor(v, off, 32);
  return v;
}
__device__ inline float wave_max(float v) {
#pragma unroll
  for (int off = 16; off; off >>= 1) v = fmaxf(v, __shfl_xor(v, off, 32));
  return v;
}

// ---------------------------------------------------------------------------
// Generic WMMA GEMM: C[M,N] = epi(A[M,K] @ W[K,N] + bias) (+ residual)
// 64x64 tile per 128-thread (4-wave) block, K stepped by 32.
// LDS holds tiles in WMMA *fragment order*: [frag][lane][16h] (+pad),
// so compute does 2x ds_load_b128 per fragment instead of 16x ds_load_u16.
// Requires M%64==0, N%64==0, K%32==0. act: 0=none, 1=relu, 2=exact gelu.
// ---------------------------------------------------------------------------
#define FRAG_LN 24   // halves per lane slot: 16 used + 8 pad (48B stride, 16B aligned)

__global__ __launch_bounds__(128) void gemm_wmma_kernel(
    const float* __restrict__ A, const float* __restrict__ W,
    const float* __restrict__ bias, const float* __restrict__ res,
    float* __restrict__ C, int M, int N, int K, int act)
{
  __shared__ __align__(16) _Float16 sAf[4][32][FRAG_LN];  // A frag per wave
  __shared__ __align__(16) _Float16 sBf[4][32][FRAG_LN];  // B frag per n-subtile

  const int tid  = threadIdx.x;
  const int lane = tid & 31;
  const int wv   = tid >> 5;
  const int nn   = lane & 15;
  const int hi   = lane >> 4;

  const int m0 = blockIdx.y * 64;
  const int n0 = blockIdx.x * 64;

  v8f acc[4];
#pragma unroll
  for (int j = 0; j < 4; ++j) acc[j] = v8f_zero();

  const int gr  = m0 + 16 * wv + nn;           // A row this thread stages
  const int gcB = n0 + 16 * wv + nn;           // B column this thread stages (j = wv)

  for (int k0 = 0; k0 < K; k0 += 32) {
    // ---- stage A fragment row: lane holds A[gr][k0 + {8hi..8hi+7, 16+8hi..}] ----
    {
      const float* ap = A + (size_t)gr * K + k0 + 8 * hi;
      v8h lo, hh;
#pragma unroll
      for (int t = 0; t < 8; ++t) lo[t] = (_Float16)ap[t];
#pragma unroll
      for (int t = 0; t < 8; ++t) hh[t] = (_Float16)ap[16 + t];
      *(v8h*)&sAf[wv][lane][0] = lo;
      *(v8h*)&sAf[wv][lane][8] = hh;
      if (k0 + 32 < K) __builtin_prefetch(ap + 32, 0, 3);
    }
    // ---- stage B fragment (j = wv): lane holds W[k0 + {8hi..8hi+7, 16+8hi..}][gcB] ----
    {
      const float* wp = W + (size_t)(k0 + 8 * hi) * N + gcB;
      v8h lo, hh;
#pragma unroll
      for (int t = 0; t < 8; ++t) lo[t] = (_Float16)wp[(size_t)t * N];
#pragma unroll
      for (int t = 0; t < 8; ++t) hh[t] = (_Float16)wp[(size_t)(16 + t) * N];
      *(v8h*)&sBf[wv][lane][0] = lo;
      *(v8h*)&sBf[wv][lane][8] = hh;
      if (k0 + 32 < K) __builtin_prefetch(wp + (size_t)32 * N, 0, 3);
    }
    __syncthreads();

    v16h af = load_frag(&sAf[wv][lane][0]);
#pragma unroll
    for (int j = 0; j < 4; ++j) {
      v16h bf = load_frag(&sBf[j][lane][0]);
      acc[j] = wmma16x16x32(af, bf, acc[j]);
    }
    __syncthreads();
  }

  // epilogue: C/D layout -> VGPR r holds row (r + 8*hi), col = lane&15
#pragma unroll
  for (int j = 0; j < 4; ++j) {
    int gc = n0 + 16 * j + nn;
    float bv = bias ? bias[gc] : 0.f;
#pragma unroll
    for (int r = 0; r < 8; ++r) {
      int grr = m0 + 16 * wv + r + 8 * hi;
      float v = acc[j][r] + bv;
      if (act == 1)      v = fmaxf(v, 0.f);
      else if (act == 2) v = 0.5f * v * (1.f + erff(v * 0.70710678118654752f));
      if (res) v += res[(size_t)grr * N + gc];
      C[(size_t)grr * N + gc] = v;
    }
  }
}

// ---------------------------------------------------------------------------
// BatchNorm stats: one block per (group, column). mean/var indexed g*N+c.
// ---------------------------------------------------------------------------
__global__ __launch_bounds__(256) void bn_stats_kernel(
    const float* __restrict__ x, float* __restrict__ mean,
    float* __restrict__ var, int R, int N)
{
  int c = blockIdx.x % N;
  int g = blockIdx.x / N;
  const float* base = x + (size_t)g * R * N + c;
  float s = 0.f, q = 0.f;
  for (int r = threadIdx.x; r < R; r += 256) {
    float v = base[(size_t)r * N];
    s += v; q += v * v;
  }
  __shared__ float ss[256], sq[256];
  ss[threadIdx.x] = s; sq[threadIdx.x] = q;
  __syncthreads();
  for (int off = 128; off; off >>= 1) {
    if (threadIdx.x < off) {
      ss[threadIdx.x] += ss[threadIdx.x + off];
      sq[threadIdx.x] += sq[threadIdx.x + off];
    }
    __syncthreads();
  }
  if (threadIdx.x == 0) {
    float m = ss[0] / (float)R;
    mean[blockIdx.x] = m;
    var[blockIdx.x]  = sq[0] / (float)R - m * m;
  }
}

__global__ void bn_apply_relu_kernel(
    const float* __restrict__ x, float* __restrict__ y,
    const float* __restrict__ mean, const float* __restrict__ var,
    const float* __restrict__ gamma, const float* __restrict__ beta,
    int R, int N, size_t total)
{
  size_t i = (size_t)blockIdx.x * 256 + threadIdx.x;
  if (i >= total) return;
  int c = (int)(i % N);
  int g = (int)(i / ((size_t)R * N));
  float m = mean[g * N + c];
  float v = (x[i] - m) * rsqrtf(var[g * N + c] + EPSV) * gamma[c] + beta[c];
  y[i] = fmaxf(v, 0.f);
}

// ---------------------------------------------------------------------------
// Visible index scan: per batch row, exclusive scan of !mask (512 -> 128 vis)
// ---------------------------------------------------------------------------
__global__ __launch_bounds__(512) void visidx_kernel(
    const unsigned char* __restrict__ mask, int* __restrict__ vis_idx,
    int* __restrict__ slot)
{
  __shared__ int sc[512];
  int b = blockIdx.x, p = threadIdx.x;
  int vis = mask[b * NP + p] ? 0 : 1;
  sc[p] = vis;
  __syncthreads();
  for (int off = 1; off < 512; off <<= 1) {
    int v = sc[p];
    int add = (p >= off) ? sc[p - off] : 0;
    __syncthreads();
    sc[p] = v + add;
    __syncthreads();
  }
  int excl = sc[p] - vis;
  slot[b * NP + p] = vis ? excl : -1;
  if (vis) vis_idx[b * NV + excl] = p;
}

__global__ void gather_vis_kernel(const float* __restrict__ x,
                                  const int* __restrict__ vis_idx,
                                  float* __restrict__ visp)
{
  size_t i = (size_t)blockIdx.x * 256 + threadIdx.x;
  if (i >= (size_t)NVT * NCIN) return;
  int c = (int)(i & 255);
  int v = (int)((i >> 8) & 127);
  int b = (int)(i >> 15);
  int p = vis_idx[b * NV + v];
  visp[i] = x[((size_t)(b * NP + p)) * NCIN + c];
}

__global__ void posadd_kernel(float* __restrict__ ev2,
                              const float* __restrict__ pos,
                              const int* __restrict__ vis_idx)
{
  size_t i = (size_t)blockIdx.x * 256 + threadIdx.x;
  if (i >= (size_t)NVT * NEND) return;
  int c = (int)(i & 127);
  int v = (int)((i >> 7) & 127);
  int b = (int)(i >> 14);
  int p = vis_idx[b * NV + v];
  ev2[i] += pos[p * NEND + c];
}

__global__ void build_full_kernel(const float* __restrict__ xv,
                                  const int* __restrict__ slot,
                                  const float* __restrict__ mask_token,
                                  const float* __restrict__ dec_pos,
                                  float* __restrict__ full)
{
  size_t i = (size_t)blockIdx.x * 256 + threadIdx.x;
  if (i >= (size_t)NTOK * NDD) return;
  int c = (int)(i & 255);
  int p = (int)((i >> 8) & 511);
  int b = (int)(i >> 17);
  int s = slot[b * NP + p];
  float v = (s >= 0) ? xv[((size_t)(b * NV + s)) * NDD + c] : mask_token[c];
  full[i] = v + dec_pos[p * NDD + c];
}

// ---------------------------------------------------------------------------
// LayerNorm over D=256: one wave per token (8 f32/lane), wave32 shfl tree
// ---------------------------------------------------------------------------
__global__ __launch_bounds__(256) void ln_kernel(
    const float* __restrict__ x, const float* __restrict__ g,
    const float* __restrict__ bb, float* __restrict__ y)
{
  int lane = threadIdx.x & 31;
  int wv   = threadIdx.x >> 5;
  size_t tok = (size_t)blockIdx.x * 8 + wv;
  const float* xr = x + tok * NDD;
  float v[8];
  float s = 0.f;
#pragma unroll
  for (int i = 0; i < 8; ++i) { v[i] = xr[lane + 32 * i]; s += v[i]; }
  s = wave_sum(s);
  float mean = s * (1.f / NDD);
  float q = 0.f;
#pragma unroll
  for (int i = 0; i < 8; ++i) { float d = v[i] - mean; q += d * d; }
  q = wave_sum(q);
  float inv = rsqrtf(q * (1.f / NDD) + EPSV);
  float* yr = y + tok * NDD;
#pragma unroll
  for (int i = 0; i < 8; ++i) {
    int c = lane + 32 * i;
    yr[c] = (v[i] - mean) * inv * g[c] + bb[c];
  }
}

// ---------------------------------------------------------------------------
// Attention: one wave per (b, head, 16-row q tile).
// scores(16x512) via WMMA kept in LDS (f32), softmax, P(f16) @ V via WMMA.
// P fragments read from LDS as contiguous 8-half runs (ds_load_b128).
// qkv layout: (NTOK, 768) = [Q | K | V] each (NTOK, 256).
// ---------------------------------------------------------------------------
__global__ __launch_bounds__(32) void attn_kernel(
    const float* __restrict__ qkv, float* __restrict__ out)
{
  __shared__ __align__(16) float    ssc[16][NP];  // 32 KB
  __shared__ __align__(16) _Float16 sp[16][NP];   // 16 KB
  const int lane = threadIdx.x;
  const int n  = lane & 15;
  const int hi = lane >> 4;
  const int qt = blockIdx.x & 31;          // NP/16
  const int h  = (blockIdx.x >> 5) & 3;    // NHEADS
  const int b  = blockIdx.x >> 7;
  const size_t ld = 3 * NDD;               // 768

  // Q fragments: K-dim 64 => two 16x32 A fragments (contiguous f32 runs)
  const float* qbase = qkv + ((size_t)(b * NP + qt * 16 + n)) * ld + h * NDH;
  v16h aq[2];
#pragma unroll
  for (int s = 0; s < 2; ++s)
#pragma unroll
    for (int g = 0; g < 2; ++g)
#pragma unroll
      for (int t = 0; t < 8; ++t)
        aq[s][8 * g + t] = (_Float16)qbase[32 * s + 16 * hi + 8 * g + t];

  // scores = Q @ K^T * 1/sqrt(dh)
  for (int kt = 0; kt < NP / 16; ++kt) {
    const float* kbase =
        qkv + ((size_t)(b * NP + kt * 16 + n)) * ld + NDD + h * NDH;
    v8f acc = v8f_zero();
#pragma unroll
    for (int s = 0; s < 2; ++s) {
      v16h bf;
#pragma unroll
      for (int t = 0; t < 16; ++t)
        bf[t] = (_Float16)kbase[32 * s + 16 * hi + t];
      acc = wmma16x16x32(aq[s], bf, acc);
    }
#pragma unroll
    for (int r = 0; r < 8; ++r)
      ssc[r + 8 * hi][kt * 16 + n] = acc[r] * 0.125f;
  }
  __syncthreads();

  // softmax per row (16 cols/lane)
  for (int row = 0; row < 16; ++row) {
    float vals[16];
    float mx = -INFINITY;
#pragma unroll
    for (int i = 0; i < 16; ++i) {
      vals[i] = ssc[row][lane + 32 * i];
      mx = fmaxf(mx, vals[i]);
    }
    mx = wave_max(mx);
    float sum = 0.f;
#pragma unroll
    for (int i = 0; i < 16; ++i) { vals[i] = expf(vals[i] - mx); sum += vals[i]; }
    sum = wave_sum(sum);
    float inv = 1.f / sum;
#pragma unroll
    for (int i = 0; i < 16; ++i)
      sp[row][lane + 32 * i] = (_Float16)(vals[i] * inv);
  }
  __syncthreads();

  // O = P @ V  (K = 512 in 16 WMMA steps, dh = 64 => 4 n-subtiles)
  v8f oacc[4];
#pragma unroll
  for (int j = 0; j < 4; ++j) oacc[j] = v8f_zero();
  for (int s = 0; s < NP / 32; ++s) {
    // A frag from LDS: two contiguous 8-half runs -> ds_load_b128
    const _Float16* pp = &sp[n][32 * s];
    v16h af = cat8(*(const v8h*)(pp + 16 * hi), *(const v8h*)(pp + 16 * hi + 8));
    const float* vbase =
        qkv + ((size_t)(b * NP + 32 * s + 16 * hi)) * ld + 2 * NDD + h * NDH;
#pragma unroll
    for (int j = 0; j < 4; ++j) {
      v16h bf;
#pragma unroll
      for (int t = 0; t < 16; ++t)
        bf[t] = (_Float16)vbase[(size_t)t * ld + 16 * j + n];
      oacc[j] = wmma16x16x32(af, bf, oacc[j]);
    }
  }
#pragma unroll
  for (int j = 0; j < 4; ++j)
#pragma unroll
    for (int r = 0; r < 8; ++r)
      out[((size_t)(b * NP + qt * 16 + r + 8 * hi)) * NDD + h * NDH + 16 * j + n] =
          oacc[j][r];
}

// ---------------------------------------------------------------------------
// Loss: deterministic two-stage reduction (no float atomics)
// ---------------------------------------------------------------------------
__global__ __launch_bounds__(256) void loss_partial_kernel(
    const float* __restrict__ pred, const float* __restrict__ tgt,
    const unsigned char* __restrict__ mask, float* __restrict__ pnum,
    float* __restrict__ pden)
{
  int lane = threadIdx.x & 31, wv = threadIdx.x >> 5;
  size_t tok = (size_t)blockIdx.x * 8 + wv;
  const float* pr = pred + tok * NEND;
  const float* tr = tgt + tok * NEND;
  float s = 0.f;
#pragma unroll
  for (int i = 0; i < 4; ++i) {
    float d = pr[lane + 32 * i] - tr[lane + 32 * i];
    s += d * d;
  }
  s = wave_sum(s);
  __shared__ float sn[8], sd[8];
  if (lane == 0) {
    float mf = mask[tok] ? 1.f : 0.f;
    sn[wv] = (s * (1.f / NEND)) * mf;
    sd[wv] = mf;
  }
  __syncthreads();
  if (threadIdx.x == 0) {
    float a = 0.f, d = 0.f;
    for (int i = 0; i < 8; ++i) { a += sn[i]; d += sd[i]; }
    pnum[blockIdx.x] = a;
    pden[blockIdx.x] = d;
  }
}

__global__ __launch_bounds__(256) void loss_final_kernel(
    const float* __restrict__ pnum, const float* __restrict__ pden, int nblk,
    float* __restrict__ out)
{
  float a = 0.f, d = 0.f;
  for (int i = threadIdx.x; i < nblk; i += 256) { a += pnum[i]; d += pden[i]; }
  __shared__ float sa[256], sb[256];
  sa[threadIdx.x] = a; sb[threadIdx.x] = d;
  __syncthreads();
  for (int off = 128; off; off >>= 1) {
    if (threadIdx.x < off) {
      sa[threadIdx.x] += sa[threadIdx.x + off];
      sb[threadIdx.x] += sb[threadIdx.x + off];
    }
    __syncthreads();
  }
  if (threadIdx.x == 0) out[0] = sa[0] / (sb[0] + 1e-8f);
}

// ---------------------------------------------------------------------------
// Host orchestration
// ---------------------------------------------------------------------------
static inline void launch_gemm(const float* A, const float* W, const float* bias,
                               const float* res, float* C, int M, int N, int K,
                               int act, hipStream_t stream)
{
  dim3 grid(N / 64, M / 64);   // all shapes are multiples of 64
  gemm_wmma_kernel<<<grid, 128, 0, stream>>>(A, W, bias, res, C, M, N, K, act);
}

static inline size_t ew_blocks(size_t total) { return (total + 255) / 256; }

extern "C" void kernel_launch(void* const* d_in, const int* in_sizes, int n_in,
                              void* d_out, int out_size, void* d_ws, size_t ws_size,
                              hipStream_t stream)
{
  (void)in_sizes; (void)n_in; (void)out_size; (void)ws_size;

  const float* x           = (const float*)d_in[0];
  const unsigned char* msk = (const unsigned char*)d_in[1];
  const float* enc_w1  = (const float*)d_in[2];
  const float* enc_b1  = (const float*)d_in[3];
  const float* enc_g1  = (const float*)d_in[4];
  const float* enc_be1 = (const float*)d_in[5];
  const float* enc_w2  = (const float*)d_in[6];
  const float* enc_b2  = (const float*)d_in[7];
  const float* enc_g2  = (const float*)d_in[8];
  const float* enc_be2 = (const float*)d_in[9];
  const float* pos_embed    = (const float*)d_in[10];
  const float* dec_embed_w  = (const float*)d_in[11];
  const float* dec_embed_b  = (const float*)d_in[12];
  const float* mask_token   = (const float*)d_in[13];
  const float* dec_pos      = (const float*)d_in[14];
  const float* ln1_g  = (const float*)d_in[15];
  const float* ln1_b  = (const float*)d_in[16];
  const float* ain_w  = (const float*)d_in[17];
  const float* ain_b  = (const float*)d_in[18];
  const float* aout_w = (const float*)d_in[19];
  const float* aout_b = (const float*)d_in[20];
  const float* ln2_g  = (const float*)d_in[21];
  const float* ln2_b  = (const float*)d_in[22];
  const float* mlp_w1 = (const float*)d_in[23];
  const float* mlp_b1 = (const float*)d_in[24];
  const float* mlp_w2 = (const float*)d_in[25];
  const float* mlp_b2 = (const float*)d_in[26];
  const float* norm_g = (const float*)d_in[27];
  const float* norm_b = (const float*)d_in[28];
  const float* pred_w = (const float*)d_in[29];
  const float* pred_b = (const float*)d_in[30];

  float* out_loss = (float*)d_out;
  float* pred     = (float*)d_out + 1;   // (NTOK, 128)

  // ---- workspace bump allocator (256B aligned) ----
  char* wp = (char*)d_ws;
  auto alloc = [&](size_t bytes) -> void* {
    void* r = (void*)wp;
    wp += (bytes + 255) & ~(size_t)255;
    return r;
  };
  float* stat_m  = (float*)alloc((size_t)NB * NENH * sizeof(float));
  float* stat_v  = (float*)alloc((size_t)NB * NENH * sizeof(float));
  float* h1      = (float*)alloc((size_t)NTOK * NDD * sizeof(float));
  float* target  = (float*)alloc((size_t)NTOK * NEND * sizeof(float));
  int*   vis_idx = (int*)alloc((size_t)NVT * sizeof(int));
  int*   slot    = (int*)alloc((size_t)NTOK * sizeof(int));
  float* visp    = (float*)alloc((size_t)NVT * NCIN * sizeof(float));
  float* ev1     = (float*)alloc((size_t)NVT * NENH * sizeof(float));
  float* ev2     = (float*)alloc((size_t)NVT * NEND * sizeof(float));
  float* xv      = (float*)alloc((size_t)NVT * NDD * sizeof(float));
  float* full    = (float*)alloc((size_t)NTOK * NDD * sizeof(float));
  float* big     = (float*)alloc((size_t)NTOK * 1024 * sizeof(float)); // qkv & mlp hidden
  const int NLOSSB = NTOK / 8;
  float* pnum    = (float*)alloc((size_t)NLOSSB * sizeof(float));
  float* pden    = (float*)alloc((size_t)NLOSSB * sizeof(float));

  // ---- target encoder: global BN over (B,P) ----
  launch_gemm(x, enc_w1, enc_b1, nullptr, h1, NTOK, NENH, NCIN, 0, stream);
  bn_stats_kernel<<<NENH, 256, 0, stream>>>(h1, stat_m, stat_v, NTOK, NENH);
  bn_apply_relu_kernel<<<ew_blocks((size_t)NTOK * NENH), 256, 0, stream>>>(
      h1, h1, stat_m, stat_v, enc_g1, enc_be1, NTOK, NENH, (size_t)NTOK * NENH);
  launch_gemm(h1, enc_w2, enc_b2, nullptr, target, NTOK, NEND, NENH, 0, stream);
  bn_stats_kernel<<<NEND, 256, 0, stream>>>(target, stat_m, stat_v, NTOK, NEND);
  bn_apply_relu_kernel<<<ew_blocks((size_t)NTOK * NEND), 256, 0, stream>>>(
      target, target, stat_m, stat_v, enc_g2, enc_be2, NTOK, NEND,
      (size_t)NTOK * NEND);

  // ---- visible path: gather + per-sample-BN encoder ----
  visidx_kernel<<<NB, 512, 0, stream>>>(msk, vis_idx, slot);
  gather_vis_kernel<<<ew_blocks((size_t)NVT * NCIN), 256, 0, stream>>>(
      x, vis_idx, visp);
  launch_gemm(visp, enc_w1, enc_b1, nullptr, ev1, NVT, NENH, NCIN, 0, stream);
  bn_stats_kernel<<<NB * NENH, 256, 0, stream>>>(ev1, stat_m, stat_v, NV, NENH);
  bn_apply_relu_kernel<<<ew_blocks((size_t)NVT * NENH), 256, 0, stream>>>(
      ev1, ev1, stat_m, stat_v, enc_g1, enc_be1, NV, NENH, (size_t)NVT * NENH);
  launch_gemm(ev1, enc_w2, enc_b2, nullptr, ev2, NVT, NEND, NENH, 0, stream);
  bn_stats_kernel<<<NB * NEND, 256, 0, stream>>>(ev2, stat_m, stat_v, NV, NEND);
  bn_apply_relu_kernel<<<ew_blocks((size_t)NVT * NEND), 256, 0, stream>>>(
      ev2, ev2, stat_m, stat_v, enc_g2, enc_be2, NV, NEND, (size_t)NVT * NEND);
  posadd_kernel<<<ew_blocks((size_t)NVT * NEND), 256, 0, stream>>>(
      ev2, pos_embed, vis_idx);

  // ---- decoder embed + scatter with mask tokens ----
  launch_gemm(ev2, dec_embed_w, dec_embed_b, nullptr, xv, NVT, NDD, NEND, 0,
              stream);
  build_full_kernel<<<ew_blocks((size_t)NTOK * NDD), 256, 0, stream>>>(
      xv, slot, mask_token, dec_pos, full);

  // ---- 2 transformer layers ----
  const int ntokb = NTOK / 8;
  for (int l = 0; l < 2; ++l) {
    ln_kernel<<<ntokb, 256, 0, stream>>>(full, ln1_g + l * NDD, ln1_b + l * NDD,
                                         h1);
    launch_gemm(h1, ain_w + (size_t)l * NDD * 3 * NDD, ain_b + l * 3 * NDD,
                nullptr, big, NTOK, 3 * NDD, NDD, 0, stream);
    attn_kernel<<<NB * NHEADS * (NP / 16), 32, 0, stream>>>(big, h1);
    launch_gemm(h1, aout_w + (size_t)l * NDD * NDD, aout_b + l * NDD, full,
                full, NTOK, NDD, NDD, 0, stream);
    ln_kernel<<<ntokb, 256, 0, stream>>>(full, ln2_g + l * NDD, ln2_b + l * NDD,
                                         h1);
    launch_gemm(h1, mlp_w1 + (size_t)l * NDD * 4 * NDD, mlp_b1 + l * 4 * NDD,
                nullptr, big, NTOK, 4 * NDD, NDD, 2, stream);
    launch_gemm(big, mlp_w2 + (size_t)l * 4 * NDD * NDD, mlp_b2 + l * NDD, full,
                full, NTOK, NDD, 4 * NDD, 0, stream);
  }

  // ---- final norm + prediction head ----
  ln_kernel<<<ntokb, 256, 0, stream>>>(full, norm_g, norm_b, h1);
  launch_gemm(h1, pred_w, pred_b, nullptr, pred, NTOK, NEND, NDD, 0, stream);

  // ---- masked MSE loss ----
  loss_partial_kernel<<<NLOSSB, 256, 0, stream>>>(pred, target, msk, pnum, pden);
  loss_final_kernel<<<1, 256, 0, stream>>>(pnum, pden, NLOSSB, out_loss);
}
